// StandAloneSelfAttention_42331197669625
// MI455X (gfx1250) — compile-verified, matrix operand order checked by reading, not run on hardware
//
#include <hip/hip_runtime.h>

typedef __attribute__((ext_vector_type(16))) _Float16 v16h;
typedef __attribute__((ext_vector_type(8)))  _Float16 v8h;
typedef __attribute__((ext_vector_type(4)))  _Float16 v4h;
typedef __attribute__((ext_vector_type(2)))  _Float16 v2h;
typedef __attribute__((ext_vector_type(8)))  float    v8f;

#define BSZ   2
#define HDIM  64
#define WDIM  64
#define CDIM  256
#define FDIM  256
#define HEADS 8
#define HS    32
#define KSZ   7
#define NPIX  (BSZ * HDIM * WDIM)   // 8192

// ---------------------------------------------------------------------------
// Convert x (fp32, row-major [8192,256]) -> f16
// ---------------------------------------------------------------------------
__global__ void cvt_x_kernel(const float* __restrict__ x, _Float16* __restrict__ xh) {
    int i = (blockIdx.x * blockDim.x + threadIdx.x) * 4;
    float4 f = *(const float4*)(x + i);
    v4h h;
    h[0] = (_Float16)f.x; h[1] = (_Float16)f.y;
    h[2] = (_Float16)f.z; h[3] = (_Float16)f.w;
    *(v4h*)(xh + i) = h;
}

// ---------------------------------------------------------------------------
// Convert + transpose weight: WT[n][c] = (f16) W[c][n]   (256x256)
// ---------------------------------------------------------------------------
__global__ void cvt_wT_kernel(const float* __restrict__ w, _Float16* __restrict__ wt) {
    int idx = blockIdx.x * blockDim.x + threadIdx.x;   // 65536 total
    int n = idx >> 8;
    int c = idx & 255;
    wt[n * CDIM + c] = (_Float16)w[c * FDIM + n];
}

// ---------------------------------------------------------------------------
// Fused QKV GEMM:  out[proj] = xh(f16)[8192,256] * WT(f16)^T + bias, fp32 out
// Block = 128 threads (4 waves); each wave computes a 64x64 tile via 16 WMMA
// accumulators. grid = (M/128, N/128, 3).
// ---------------------------------------------------------------------------
__device__ __forceinline__ v16h load_frag16(const _Float16* rowp, int k0, int half) {
    // 16-bit A/B fragment layout (CDNA5 ISA 7.12.2):
    //  lanes 0-15 : elements 0-7 = K[k0..k0+7],   8-15 = K[k0+16..k0+23]
    //  lanes16-31 : elements 0-7 = K[k0+8..k0+15],8-15 = K[k0+24..k0+31]
    union { v16h v; v8h h[2]; } u;
    const _Float16* p = rowp + k0 + half * 8;
    u.h[0] = *(const v8h*)(p);
    u.h[1] = *(const v8h*)(p + 16);
    return u.v;
}

__global__ __launch_bounds__(128)
void qkv_gemm_kernel(const _Float16* __restrict__ xh,
                     const _Float16* __restrict__ wqt,
                     const _Float16* __restrict__ wkt,
                     const _Float16* __restrict__ wvt,
                     const float* __restrict__ bq,
                     const float* __restrict__ bk,
                     const float* __restrict__ bv,
                     float* __restrict__ qo,
                     float* __restrict__ ko,
                     float* __restrict__ vo) {
    const int proj = blockIdx.z;
    const _Float16* wt   = (proj == 0) ? wqt : (proj == 1) ? wkt : wvt;
    const float*    bias = (proj == 0) ? bq  : (proj == 1) ? bk  : bv;
    float*          out  = (proj == 0) ? qo  : (proj == 1) ? ko  : vo;

    const int wave = threadIdx.x >> 5;
    const int lane = threadIdx.x & 31;
    const int half = lane >> 4;
    const int lr   = lane & 15;

    const int mBase = blockIdx.x * 128 + (wave & 1) * 64;
    const int nBase = blockIdx.y * 128 + (wave >> 1) * 64;

    v8f acc[4][4] = {};

    for (int k0 = 0; k0 < CDIM; k0 += 32) {
        v16h a[4], bfr[4];
#pragma unroll
        for (int mi = 0; mi < 4; ++mi)
            a[mi] = load_frag16(xh + (mBase + mi * 16 + lr) * CDIM, k0, half);
#pragma unroll
        for (int ni = 0; ni < 4; ++ni)
            bfr[ni] = load_frag16(wt + (nBase + ni * 16 + lr) * CDIM, k0, half);
#pragma unroll
        for (int mi = 0; mi < 4; ++mi)
#pragma unroll
            for (int ni = 0; ni < 4; ++ni)
                acc[mi][ni] = __builtin_amdgcn_wmma_f32_16x16x32_f16(
                    false, a[mi], false, bfr[ni], (short)0, acc[mi][ni], false, false);
    }

    // Epilogue: C/D layout: VGPR r -> M = r + 8*half, N = lr.
#pragma unroll
    for (int ni = 0; ni < 4; ++ni) {
        const int col = nBase + ni * 16 + lr;
        const float bcol = bias[col];
#pragma unroll
        for (int mi = 0; mi < 4; ++mi) {
#pragma unroll
            for (int r = 0; r < 8; ++r) {
                const int row = mBase + mi * 16 + half * 8 + r;
                out[row * FDIM + col] = acc[mi][ni][r] + bcol;
            }
        }
    }
}

// ---------------------------------------------------------------------------
// Local 7x7 attention. Tile 8x4 pixels per block, halo 14x10 of k/v in LDS
// (f16). 256 threads, one (pixel, head) task each.
// ---------------------------------------------------------------------------
#define TH     8
#define TW     4
#define HALO_H (TH + KSZ - 1)        // 14
#define HALO_W (TW + KSZ - 1)        // 10
#define NPOS   (HALO_H * HALO_W)     // 140
#define KROW   264                   // 256 + 8 f16 pad (bank-conflict stagger)

__global__ __launch_bounds__(256)
void local_attn_kernel(const float* __restrict__ qb,
                       const float* __restrict__ kb,
                       const float* __restrict__ vb,
                       const float* __restrict__ emb0,
                       const float* __restrict__ emb1,
                       float* __restrict__ out) {
    __shared__ _Float16 ksh[NPOS * KROW];   // 72 KB
    __shared__ _Float16 vsh[NPOS * KROW];   // 72 KB

    const int b   = blockIdx.z;
    const int y0  = blockIdx.y * TH;
    const int x0  = blockIdx.x * TW;
    const int tid = threadIdx.x;

    // ---- cooperative halo fill (zero outside image: reference zero-pads) ----
    for (int e = tid; e < NPOS * (CDIM / 2); e += 256) {
        const int pos = e >> 7;                 // / 128 channel-pairs
        const int cp  = e & 127;
        const int iy  = pos / HALO_W;
        const int ix  = pos - iy * HALO_W;
        const int gy  = y0 + iy - (KSZ / 2);
        const int gx  = x0 + ix - (KSZ / 2);
        float2 kf = make_float2(0.f, 0.f);
        float2 vf = make_float2(0.f, 0.f);
        if (gy >= 0 && gy < HDIM && gx >= 0 && gx < WDIM) {
            const int base = (((b * HDIM + gy) * WDIM + gx) << 8) + (cp << 1);
            kf = *(const float2*)(kb + base);
            vf = *(const float2*)(vb + base);
        }
        v2h kh; kh[0] = (_Float16)kf.x; kh[1] = (_Float16)kf.y;
        v2h vh; vh[0] = (_Float16)vf.x; vh[1] = (_Float16)vf.y;
        *(v2h*)&ksh[pos * KROW + (cp << 1)] = kh;
        *(v2h*)&vsh[pos * KROW + (cp << 1)] = vh;
    }
    __syncthreads();

    // ---- one (pixel, head) per thread ----
    const int pix  = tid >> 3;          // 0..31
    const int head = tid & 7;
    const int py   = pix >> 2;          // / TW
    const int px   = pix & 3;
    const int y    = y0 + py;
    const int x    = x0 + px;

    const int qbase = (((b * HDIM + y) * WDIM + x) << 8) + head * HS;
    float q[HS];
#pragma unroll
    for (int d4 = 0; d4 < HS / 4; ++d4) {
        float4 f = *(const float4*)(qb + qbase + d4 * 4);
        q[d4 * 4 + 0] = f.x; q[d4 * 4 + 1] = f.y;
        q[d4 * 4 + 2] = f.z; q[d4 * 4 + 3] = f.w;
    }

    // qe[a] = q . e(:,a): heads 0-3 use emb0 (depends on di); 4-7 use emb1 (dj)
    const bool  useRow = (head < 4);
    const float* eptr  = useRow ? (emb0 + head * HS * KSZ)
                                : (emb1 + (head - 4) * HS * KSZ);
    float qe[KSZ] = {0.f, 0.f, 0.f, 0.f, 0.f, 0.f, 0.f};
#pragma unroll
    for (int d = 0; d < HS; ++d)
#pragma unroll
        for (int a = 0; a < KSZ; ++a)
            qe[a] += q[d] * eptr[d * KSZ + a];

    // ---- scores ----
    float sc[KSZ * KSZ];
    float mx = -3.0e38f;
#pragma unroll
    for (int di = 0; di < KSZ; ++di) {
#pragma unroll
        for (int dj = 0; dj < KSZ; ++dj) {
            const int pos = (py + di) * HALO_W + (px + dj);
            const v8h* kp = (const v8h*)&ksh[pos * KROW + head * HS];
            float s = useRow ? qe[di] : qe[dj];
#pragma unroll
            for (int c8 = 0; c8 < 4; ++c8) {
                v8h kv = kp[c8];
#pragma unroll
                for (int j = 0; j < 8; ++j)
                    s += q[c8 * 8 + j] * (float)kv[j];
            }
            sc[di * KSZ + dj] = s;
            mx = fmaxf(mx, s);
        }
    }

    float denom = 0.f;
#pragma unroll
    for (int p = 0; p < KSZ * KSZ; ++p) {
        float e = __expf(sc[p] - mx);
        sc[p] = e;
        denom += e;
    }
    const float inv = 1.0f / denom;

    // ---- weighted value sum ----
    float o[HS] = {};
#pragma unroll
    for (int di = 0; di < KSZ; ++di) {
#pragma unroll
        for (int dj = 0; dj < KSZ; ++dj) {
            const int pos = (py + di) * HALO_W + (px + dj);
            const v8h* vp = (const v8h*)&vsh[pos * KROW + head * HS];
            const float w = sc[di * KSZ + dj];
#pragma unroll
            for (int c8 = 0; c8 < 4; ++c8) {
                v8h vv = vp[c8];
#pragma unroll
                for (int j = 0; j < 8; ++j)
                    o[c8 * 8 + j] += w * (float)vv[j];
            }
        }
    }

#pragma unroll
    for (int d4 = 0; d4 < HS / 4; ++d4) {
        float4 f;
        f.x = o[d4 * 4 + 0] * inv; f.y = o[d4 * 4 + 1] * inv;
        f.z = o[d4 * 4 + 2] * inv; f.w = o[d4 * 4 + 3] * inv;
        *(float4*)(out + qbase + d4 * 4) = f;
    }
}

// ---------------------------------------------------------------------------
// Launch
// ---------------------------------------------------------------------------
extern "C" void kernel_launch(void* const* d_in, const int* in_sizes, int n_in,
                              void* d_out, int out_size, void* d_ws, size_t ws_size,
                              hipStream_t stream) {
    (void)in_sizes; (void)n_in; (void)out_size; (void)ws_size;

    const float* x    = (const float*)d_in[0];
    const float* Wq   = (const float*)d_in[1];
    const float* bq   = (const float*)d_in[2];
    const float* Wk   = (const float*)d_in[3];
    const float* bk   = (const float*)d_in[4];
    const float* Wv   = (const float*)d_in[5];
    const float* bv   = (const float*)d_in[6];
    const float* emb0 = (const float*)d_in[7];
    const float* emb1 = (const float*)d_in[8];
    float* out = (float*)d_out;

    char* ws = (char*)d_ws;
    _Float16* xh  = (_Float16*)(ws);                        // 4 MB  [8192,256] f16
    _Float16* wqt = (_Float16*)(ws + (4u << 20));           // 128 KB each (W^T f16)
    _Float16* wkt = wqt + CDIM * FDIM;
    _Float16* wvt = wkt + CDIM * FDIM;
    float* qb = (float*)(ws + (5u  << 20));                 // 8 MB fp32
    float* kb = (float*)(ws + (13u << 20));                 // 8 MB fp32
    float* vb = (float*)(ws + (21u << 20));                 // 8 MB fp32

    cvt_x_kernel<<<(NPIX * CDIM) / (4 * 256), 256, 0, stream>>>(x, xh);
    cvt_wT_kernel<<<(CDIM * FDIM) / 256, 256, 0, stream>>>(Wq, wqt);
    cvt_wT_kernel<<<(CDIM * FDIM) / 256, 256, 0, stream>>>(Wk, wkt);
    cvt_wT_kernel<<<(CDIM * FDIM) / 256, 256, 0, stream>>>(Wv, wvt);

    qkv_gemm_kernel<<<dim3(NPIX / 128, FDIM / 128, 3), 128, 0, stream>>>(
        xh, wqt, wkt, wvt, bq, bk, bv, qb, kb, vb);

    local_attn_kernel<<<dim3(WDIM / TW, HDIM / TH, BSZ), 256, 0, stream>>>(
        qb, kb, vb, emb0, emb1, out);
}
